// MultiscaleFlow_63548336111935
// MI455X (gfx1250) — matrix operand order repaired and checked by reading, not compiled, not run on hardware
//
#include <hip/hip_runtime.h>
#include <hip/hip_bf16.h>
#include <math.h>

// MI455X / gfx1250, wave32. Distance GEMM on v_wmma_f32_16x16x32_f16 with
// double-buffered async global->LDS anchor staging (ASYNCcnt pipeline).

typedef __attribute__((ext_vector_type(16))) _Float16 v16h;
typedef __attribute__((ext_vector_type(8)))  _Float16 v8h;
typedef __attribute__((ext_vector_type(8)))  float    v8f;
typedef __attribute__((vector_size(4 * sizeof(int)))) int v4i;   // matches builtin proto

#define D_DIM   256
#define WG_ROWS 64          // rows per workgroup (4 waves x 16)
#define CHUNK_A 64          // anchors staged per LDS chunk
#define LDS_STR 272         // 256 + 16 halfs pad -> 544 B row stride (32B aligned)
#define NEG_LOG098 0.0202027345f   // -log(0.98)

#define AS1 __attribute__((address_space(1)))
#define AS3 __attribute__((address_space(3)))

#if defined(__HIP_DEVICE_COMPILE__) && __has_builtin(__builtin_amdgcn_global_load_async_to_lds_b128)
#define HAVE_ASYNC_LDS 1
#else
#define HAVE_ASYNC_LDS 0
#endif

// ---------------- anchor prep: f16 copy + ||a_k||^2 ----------------
__global__ void k_anchor_prep(const float* __restrict__ anchors,
                              _Float16* __restrict__ ah,
                              float* __restrict__ anorm) {
  const int k = blockIdx.x, d = threadIdx.x;
  const float a = anchors[(size_t)k * D_DIM + d];
  ah[(size_t)k * D_DIM + d] = (_Float16)a;
  __shared__ float red[D_DIM];
  red[d] = a * a;
  __syncthreads();
  for (int s = 128; s > 0; s >>= 1) { if (d < s) red[d] += red[d + s]; __syncthreads(); }
  if (d == 0) anorm[k] = red[0];
}

// ---------------- row-independent logp constants + log_softmax(mix) ----------------
__global__ void k_const_lsm(const float* __restrict__ a1w, const float* __restrict__ a2w,
                            const float* __restrict__ shift,
                            const float* __restrict__ bmin, const float* __restrict__ bmax,
                            const float* __restrict__ mix, int K,
                            float* __restrict__ lsm, float* __restrict__ constv) {
  const int t = threadIdx.x;            // blockDim.x == 1024
  __shared__ float red[1024];
  float term = 0.f;
  if (t < D_DIM) {
    const float sc = 1.f / (1.f + expf(-shift[t])) * 0.98f + 0.01f;
    term = -a1w[t] - a2w[t] - logf(bmax[t] - bmin[t]) - log1pf(-sc);
  }
  red[t] = term; __syncthreads();
  for (int s = 512; s > 0; s >>= 1) { if (t < s) red[t] += red[t + s]; __syncthreads(); }
  const float cst = red[0] + (float)D_DIM * NEG_LOG098;  // + D*log(hi-lo), hi-lo = 1/0.98
  __syncthreads();

  const float m = (t < K) ? mix[t] : -3.4e38f;
  red[t] = m; __syncthreads();
  for (int s = 512; s > 0; s >>= 1) { if (t < s) red[t] = fmaxf(red[t], red[t + s]); __syncthreads(); }
  const float mx = red[0]; __syncthreads();
  red[t] = (t < K) ? expf(m - mx) : 0.f; __syncthreads();
  for (int s = 512; s > 0; s >>= 1) { if (t < s) red[t] += red[t + s]; __syncthreads(); }
  const float lz = mx + logf(red[0]);
  if (t < K) lsm[t] = m - lz;
  if (t == 0) constv[0] = cst;
}

// ---------------- elementwise chain + per-row sigmoid logdet ----------------
// post-box value collapses to sigmoid(y)*(bmax-bmin)+bmin  (algebraic identity)
__global__ void k_rows(const float* __restrict__ x,
                       const float* __restrict__ a1b, const float* __restrict__ a1w,
                       const float* __restrict__ bmin, const float* __restrict__ bmax,
                       float* __restrict__ xbox, _Float16* __restrict__ xh,
                       float* __restrict__ rowld) {
  const int row = blockIdx.x, d = threadIdx.x;
  const size_t o = (size_t)row * D_DIM + d;
  const float y  = (x[o] + a1b[d]) * expf(a1w[d]);
  const float s  = 1.f / (1.f + expf(-y));
  const float ay = fabsf(y);
  const float ld = -ay - 2.f * log1pf(expf(-ay)) + NEG_LOG098;  // logsig(y)+logsig(-y)-log(1-2a)
  const float xb = s * (bmax[d] - bmin[d]) + bmin[d];
  xbox[o] = xb;
  xh[o]   = (_Float16)xb;
  __shared__ float red[D_DIM];
  red[d] = ld; __syncthreads();
  for (int s2 = 128; s2 > 0; s2 >>= 1) { if (d < s2) red[d] += red[d + s2]; __syncthreads(); }
  if (d == 0) rowld[row] = red[0];
}

// --- stage one 64-anchor chunk (32 KB) into an LDS buffer; 2 threads/row ---
__device__ __forceinline__ void stage_chunk(const _Float16* __restrict__ ah,
                                            _Float16* __restrict__ sbuf,
                                            int c0, int tid) {
  const int r = tid >> 1, h2 = tid & 1;
  const char* g = (const char*)(ah + (size_t)(c0 + r) * D_DIM + h2 * 128);
  char*       l = (char*)(sbuf + r * LDS_STR + h2 * 128);
#if HAVE_ASYNC_LDS
#pragma unroll
  for (int i = 0; i < 16; ++i)
    __builtin_amdgcn_global_load_async_to_lds_b128(
        (AS1 v4i*)(g + i * 16), (AS3 v4i*)(l + i * 16), 0, 0);
#else
#pragma unroll
  for (int i = 0; i < 16; ++i)
    ((uint4*)l)[i] = ((const uint4*)g)[i];
#endif
}

__device__ __forceinline__ void wait_async_le16() {
#if HAVE_ASYNC_LDS
#if __has_builtin(__builtin_amdgcn_s_wait_asynccnt)
  __builtin_amdgcn_s_wait_asynccnt(16);
#else
  asm volatile("s_wait_asynccnt 0x10" ::: "memory");
#endif
#endif
}
__device__ __forceinline__ void wait_async_0() {
#if HAVE_ASYNC_LDS
#if __has_builtin(__builtin_amdgcn_s_wait_asynccnt)
  __builtin_amdgcn_s_wait_asynccnt(0);
#else
  asm volatile("s_wait_asynccnt 0x0" ::: "memory");
#endif
#endif
}

// ---------------- WMMA distance GEMM + argmin over K anchors ----------------
__global__ __launch_bounds__(128)
void k_gemm_argmin(const _Float16* __restrict__ xh,
                   const _Float16* __restrict__ ah,
                   const float* __restrict__ anorm,
                   int K, int* __restrict__ idx_out) {
  __shared__ _Float16 sB[2][CHUNK_A * LDS_STR];   // 2 x 34 KB double buffer
  const int tid  = threadIdx.x;
  const int lane = tid & 31;
  const int wave = tid >> 5;
  const int half = lane >> 4;                     // lane half selects K sub-block
  const int l16  = lane & 15;
  const int rowbase = blockIdx.x * WG_ROWS + wave * 16;

  // Preload all A fragments for this wave's 16 rows (16x256 f16) into VGPRs.
  // A 16x32 layout: lane<16 row=l16 holds K = kk*32 + 0..15;
  //                 lane>=16 same row holds K = kk*32 + 16..31.
  const _Float16* xrow = xh + (size_t)(rowbase + l16) * D_DIM;
  v16h afrag[8];
#pragma unroll
  for (int kk = 0; kk < 8; ++kk) {
    const v8h lo = *(const v8h*)(xrow + kk * 32 + half * 16);
    const v8h hi = *(const v8h*)(xrow + kk * 32 + half * 16 + 8);
    v16h a;
#pragma unroll
    for (int i = 0; i < 8; ++i) { a[i] = lo[i]; a[8 + i] = hi[i]; }
    afrag[kk] = a;
  }

  float bestv[8]; int besti[8];
#pragma unroll
  for (int g = 0; g < 8; ++g) { bestv[g] = 3.4e38f; besti[g] = 0; }

  const int nch = K / CHUNK_A;
  stage_chunk(ah, sB[0], 0, tid);                 // prologue: chunk 0 -> buf 0

  for (int ci = 0; ci < nch; ++ci) {
    const int p = ci & 1;
    const bool more = (ci + 1) < nch;
    if (more) stage_chunk(ah, sB[1 - p], (ci + 1) * CHUNK_A, tid);  // prefetch next
    if (more) wait_async_le16(); else wait_async_0();               // buf p landed (in-order)
    __syncthreads();                               // all waves' async + prior reads done

#pragma unroll
    for (int t = 0; t < CHUNK_A / 16; ++t) {
      // B 32x16 layout: lane<16 holds K=kk*32+0..15 of column N=l16; lane>=16 K=+16..31
      const int nloc = t * 16 + l16;
      const _Float16* brow = sB[p] + nloc * LDS_STR + half * 16;
      v8f c = {};
#pragma unroll
      for (int kk = 0; kk < 8; ++kk) {
        const v16h b = *(const v16h*)(brow + kk * 32);
        c = __builtin_amdgcn_wmma_f32_16x16x32_f16(false, afrag[kk], false, b,
                                                   (short)0, c, false, false);
      }
      const int   ng = ci * CHUNK_A + nloc;
      const float an = anorm[ng];
#pragma unroll
      for (int g = 0; g < 8; ++g) {          // slot g: lanes 0-15 -> row g, lanes 16-31 -> row 8+g
        const float sc = an - 2.f * c[g];    // ||x||^2 dropped: constant per row
        if (sc < bestv[g]) { bestv[g] = sc; besti[g] = ng; }
      }
    }
    __syncthreads();                               // reads of buf p done before it is refilled
  }

  // argmin across the 16 lanes of each half (first-min index tie-break)
#pragma unroll
  for (int g = 0; g < 8; ++g) {
    for (int off = 8; off >= 1; off >>= 1) {
      const float ov = __shfl_xor(bestv[g], off, 16);
      const int   oi = __shfl_xor(besti[g], off, 16);
      if (ov < bestv[g] || (ov == bestv[g] && oi < besti[g])) { bestv[g] = ov; besti[g] = oi; }
    }
    if (l16 == 0) idx_out[rowbase + half * 8 + g] = besti[g];
  }
}

// ---------------- gather + shift/recenter + actnorm2 + logp ----------------
__global__ void k_final(const float* __restrict__ xbox,
                        const float* __restrict__ anchors,
                        const float* __restrict__ a2b, const float* __restrict__ a2w,
                        const float* __restrict__ shift,
                        const float* __restrict__ rowld,
                        const float* __restrict__ lsm,
                        const float* __restrict__ constv,
                        const float* __restrict__ logp_in,
                        const int* __restrict__ idx,
                        float* __restrict__ xout, float* __restrict__ logp_out) {
  const int row = blockIdx.x, d = threadIdx.x;
  const int k = idx[row];
  const size_t o = (size_t)row * D_DIM + d;
  const float sc = 1.f / (1.f + expf(-shift[d])) * 0.98f + 0.01f;
  const float v  = (1.f - sc) * (xbox[o] - anchors[(size_t)k * D_DIM + d]);
  xout[o] = (v + a2b[d]) * expf(a2w[d]);
  if (d == 0) logp_out[row] = logp_in[row] + constv[0] - rowld[row] - lsm[k];
}

extern "C" void kernel_launch(void* const* d_in, const int* in_sizes, int n_in,
                              void* d_out, int out_size, void* d_ws, size_t ws_size,
                              hipStream_t stream) {
  (void)n_in; (void)out_size; (void)ws_size;
  const float* x      = (const float*)d_in[0];
  const float* logp   = (const float*)d_in[1];
  const float* a1b    = (const float*)d_in[2];
  const float* a1w    = (const float*)d_in[3];
  const float* a2b    = (const float*)d_in[4];
  const float* a2w    = (const float*)d_in[5];
  const float* shift  = (const float*)d_in[6];
  const float* anch   = (const float*)d_in[7];
  const float* mix    = (const float*)d_in[8];
  const float* bmin   = (const float*)d_in[9];
  const float* bmax   = (const float*)d_in[10];

  const int B = in_sizes[0] / D_DIM;     // 32768
  const int K = in_sizes[8];             // 1024

  // workspace carve-up (256B aligned sections)
  char* ws = (char*)d_ws; size_t off = 0;
  auto carve = [&](size_t bytes) { char* p = ws + off; off = (off + bytes + 255) & ~(size_t)255; return p; };
  _Float16* xh    = (_Float16*)carve((size_t)B * D_DIM * sizeof(_Float16));
  _Float16* ah    = (_Float16*)carve((size_t)K * D_DIM * sizeof(_Float16));
  float*    xbox  = (float*)   carve((size_t)B * D_DIM * sizeof(float));
  float*    rowld = (float*)   carve((size_t)B * sizeof(float));
  float*    anorm = (float*)   carve((size_t)K * sizeof(float));
  float*    lsm   = (float*)   carve((size_t)K * sizeof(float));
  float*    cnst  = (float*)   carve(sizeof(float));

  float* xout     = (float*)d_out;
  float* logp_out = xout + (size_t)B * D_DIM;
  int*   idx_out  = (int*)(logp_out + B);

  k_anchor_prep<<<K, D_DIM, 0, stream>>>(anch, ah, anorm);
  k_const_lsm <<<1, 1024, 0, stream>>>(a1w, a2w, shift, bmin, bmax, mix, K, lsm, cnst);
  k_rows      <<<B, D_DIM, 0, stream>>>(x, a1b, a1w, bmin, bmax, xbox, xh, rowld);
  k_gemm_argmin<<<B / WG_ROWS, 128, 0, stream>>>(xh, ah, anorm, K, idx_out);
  k_final     <<<B, D_DIM, 0, stream>>>(xbox, anch, a2b, a2w, shift, rowld, lsm, cnst,
                                        logp, idx_out, xout, logp_out);
}
